// EnhancedMoMGRUCell_89515708383668
// MI455X (gfx1250) — matrix operand order repaired and checked by reading, not compiled, not run on hardware
//
#include <hip/hip_runtime.h>
#include <hip/hip_bf16.h>

#define B_   1024
#define IN_  512
#define H_   1024
#define M_   4
#define S_   128
#define D_   64
#define RH_  32
#define TSTR 65   // padded LDS row stride -> conflict-free column reads

typedef __attribute__((ext_vector_type(16))) __bf16 v16bf;
typedef __attribute__((ext_vector_type(8)))  float  v8f;
typedef __attribute__((ext_vector_type(4)))  int    v4i;

union F16frag { v16bf v; v4i q[2]; };

__device__ __forceinline__ __bf16 f2bf(float f) {
  union { float f; unsigned u; } c; c.f = f;
  unsigned r = c.u + 0x7FFFu + ((c.u >> 16) & 1u);   // round-to-nearest-even
  unsigned short hbits = (unsigned short)(r >> 16);
  __bf16 o; __builtin_memcpy(&o, &hbits, 2); return o;
}
__device__ __forceinline__ float sigm(float x) { return 1.0f / (1.0f + __expf(-x)); }

// ---------------------------------------------------------------------------
// C[B x N] (+)= A[B x K] @ B[K x N]; A bf16 row-major, B pre-transposed bf16
// Bt[N][ldk] (koff selects a K-slice of the original weight).
// One wave owns a (16*MT) x (16*NT) tile of C. Fragments are double-buffered:
// loads for k+32 are issued while WMMAs for k execute (K % 64 == 0 assumed).
// ---------------------------------------------------------------------------
template<int MT, int NT, bool ACC>
__global__ __launch_bounds__(NT == 1 ? 32 : 128)
void gemm_bf16_wmma(const __bf16* __restrict__ A, int lda,
                    const __bf16* __restrict__ Bt, int ldk, int koff,
                    float* __restrict__ C, int ldc,
                    const float* __restrict__ bias, int K)
{
  const int lane = threadIdx.x & 31;
  const int wave = threadIdx.x >> 5;
  const int m0   = blockIdx.y * (16 * MT);
  const int n0   = (NT == 1) ? blockIdx.x * 16
                             : (blockIdx.x * 4 + wave) * (16 * NT);
  const int r    = lane & 15;
  const int hi   = lane >> 4;
  const int ka   = hi * 8;    // A: K sub-offset per ISA 16x32 bf16 layout
  const int kb   = hi * 8;    // B: K sub-offset per ISA 32x16 bf16 layout

  const __bf16* arow[MT];
  #pragma unroll
  for (int mi = 0; mi < MT; mi++)
    arow[mi] = A + (size_t)(m0 + mi * 16 + r) * lda + ka;
  const __bf16* bcol[NT];
  #pragma unroll
  for (int t = 0; t < NT; t++)
    bcol[t] = Bt + (size_t)(n0 + t * 16 + r) * ldk + koff + kb;

  v8f zero = {};
  v8f acc[MT][NT];
  #pragma unroll
  for (int mi = 0; mi < MT; mi++)
    #pragma unroll
    for (int t = 0; t < NT; t++) acc[mi][t] = zero;

  F16frag af0[MT], bf0[NT], af1[MT], bf1[NT];

  auto loadf = [&](F16frag (&af)[MT], F16frag (&bf)[NT], int k) {
    #pragma unroll
    for (int mi = 0; mi < MT; mi++) {
      af[mi].q[0] = *(const v4i*)(arow[mi] + k);        // K = base+0..7
      af[mi].q[1] = *(const v4i*)(arow[mi] + k + 16);   // K = base+16..23
    }
    #pragma unroll
    for (int t = 0; t < NT; t++) {
      bf[t].q[0] = *(const v4i*)(bcol[t] + k);
      bf[t].q[1] = *(const v4i*)(bcol[t] + k + 16);
    }
  };
  auto mmaf = [&](F16frag (&af)[MT], F16frag (&bf)[NT]) {
    #pragma unroll
    for (int mi = 0; mi < MT; mi++)
      #pragma unroll
      for (int t = 0; t < NT; t++)
        acc[mi][t] = __builtin_amdgcn_wmma_f32_16x16x32_bf16(
            false, af[mi].v, false, bf[t].v, (short)0, acc[mi][t], false, false);
  };

  // software pipeline, K % 64 == 0
  loadf(af0, bf0, 0);
  int k = 0;
  for (; k + 64 < K; k += 64) {
    loadf(af1, bf1, k + 32);
    __builtin_prefetch(arow[0] + k + 256, 0, 1);        // global_prefetch_b8
    mmaf(af0, bf0);
    loadf(af0, bf0, k + 64);
    mmaf(af1, bf1);
  }
  loadf(af1, bf1, k + 32);
  mmaf(af0, bf0);
  mmaf(af1, bf1);

  #pragma unroll
  for (int mi = 0; mi < MT; mi++) {
    const int mbase = m0 + mi * 16 + hi * 8;
    #pragma unroll
    for (int t = 0; t < NT; t++) {
      const int col = n0 + t * 16 + r;
      const float bia = ACC ? 0.0f : bias[col];
      #pragma unroll
      for (int i = 0; i < 8; i++) {
        float* cp = C + (size_t)(mbase + i) * ldc + col;
        float v = acc[mi][t][i] + bia;
        if (ACC) v += *cp;
        *cp = v;
      }
    }
  }
}

// ---------------------------------------------------------------------------
// fp32 -> bf16, same layout (n multiple of 4)
__global__ void cvt_bf16_kernel(const float* __restrict__ in, __bf16* __restrict__ out, int n)
{
  int i = (blockIdx.x * blockDim.x + threadIdx.x) * 4;
  if (i >= n) return;
  float4 v = *(const float4*)(in + i);
  out[i]   = f2bf(v.x); out[i+1] = f2bf(v.y);
  out[i+2] = f2bf(v.z); out[i+3] = f2bf(v.w);
}

// fp32 W[K][N] -> bf16 Wt[N][K], LDS-tiled transpose (K,N multiples of 32)
__global__ __launch_bounds__(256)
void cvtT_bf16_kernel(const float* __restrict__ W, __bf16* __restrict__ Wt, int K, int N)
{
  __shared__ float tile[32][33];
  const int tx = threadIdx.x;
  const int ty = threadIdx.y;
  const int k0 = blockIdx.y * 32;
  const int n0 = blockIdx.x * 32;
  #pragma unroll
  for (int j = ty; j < 32; j += 8)
    tile[j][tx] = W[(size_t)(k0 + j) * N + (n0 + tx)];
  __syncthreads();
  #pragma unroll
  for (int j = ty; j < 32; j += 8)
    Wt[(size_t)(n0 + j) * K + (k0 + tx)] = f2bf(tile[tx][j]);
}

__global__ void zero_kernel(float* __restrict__ p, int n) {
  int i = blockIdx.x * blockDim.x + threadIdx.x;
  if (i < n) p[i] = 0.0f;
}

// router: logits = relu(L1) @ Wr2 + br2 ; softmax over M
__global__ void router_kernel(const float* __restrict__ L1, const float* __restrict__ Wr2,
                              const float* __restrict__ br2, float* __restrict__ rw)
{
  int b = blockIdx.x * blockDim.x + threadIdx.x;
  if (b >= B_) return;
  float lg[M_];
  #pragma unroll
  for (int mm = 0; mm < M_; mm++) lg[mm] = br2[mm];
  const float* lp = L1 + (size_t)b * RH_;
  for (int j = 0; j < RH_; j++) {
    float hv = fmaxf(lp[j], 0.0f);
    #pragma unroll
    for (int mm = 0; mm < M_; mm++) lg[mm] += hv * Wr2[j * M_ + mm];
  }
  float mx = lg[0];
  #pragma unroll
  for (int mm = 1; mm < M_; mm++) mx = fmaxf(mx, lg[mm]);
  float s = 0.0f;
  #pragma unroll
  for (int mm = 0; mm < M_; mm++) { lg[mm] = __expf(lg[mm] - mx); s += lg[mm]; }
  float inv = 1.0f / s;
  #pragma unroll
  for (int mm = 0; mm < M_; mm++) rw[(size_t)b * M_ + mm] = lg[mm] * inv;
}

// Attention read per (b,m): 32KB tile staged in LDS, softmax over S,
// router-weighted read accumulated into ctx via f32 global atomics.
__global__ __launch_bounds__(128)
void read_ctx_kernel(const float* __restrict__ mem, const float* __restrict__ q,
                     const float* __restrict__ rw, float* __restrict__ ctx)
{
  __shared__ float tile[S_ * TSTR];
  __shared__ float sc[S_];
  __shared__ float red[S_];
  __shared__ float qs[D_];
  const int t = threadIdx.x;
  const int b = blockIdx.y;
  const int m = blockIdx.x;
  const float* mp = mem + (((size_t)b * M_ + m) * S_) * D_;
  if (t < D_) qs[t] = q[(size_t)b * D_ + t];
  __syncthreads();
  const float* rowp = mp + (size_t)t * D_;
  float* trow = tile + t * TSTR;
  float acc = 0.0f;
  #pragma unroll
  for (int d = 0; d < D_; d += 4) {
    float4 v = *(const float4*)(rowp + d);
    trow[d] = v.x; trow[d+1] = v.y; trow[d+2] = v.z; trow[d+3] = v.w;
    acc += v.x*qs[d] + v.y*qs[d+1] + v.z*qs[d+2] + v.w*qs[d+3];
  }
  float sco = acc * 0.125f;
  sc[t] = sco; red[t] = sco;
  __syncthreads();
  for (int off = 64; off > 0; off >>= 1) {
    if (t < off) red[t] = fmaxf(red[t], red[t + off]);
    __syncthreads();
  }
  float mx = red[0];
  __syncthreads();
  float p = __expf(sco - mx);
  sc[t] = p; red[t] = p;
  __syncthreads();
  for (int off = 64; off > 0; off >>= 1) {
    if (t < off) red[t] += red[t + off];
    __syncthreads();
  }
  float inv = 1.0f / red[0];
  __syncthreads();
  if (t < D_) {
    float r = 0.0f;
    #pragma unroll 4
    for (int s = 0; s < S_; s++) r += sc[s] * tile[s * TSTR + t];
    atomicAdd(&ctx[(size_t)b * D_ + t], rw[(size_t)b * M_ + m] * (r * inv));
  }
}

// r/z gates; emits r*h directly as bf16 (GEMM operand) + z in f32
__global__ void gru_gates_kernel(const float* __restrict__ A1, const float* __restrict__ A2,
                                 const float* __restrict__ h, __bf16* __restrict__ rhbf,
                                 float* __restrict__ zb, int n)
{
  int i = blockIdx.x * blockDim.x + threadIdx.x;
  if (i >= n) return;
  float r = sigm(A1[i]);
  float z = sigm(A2[i]);
  rhbf[i] = f2bf(r * h[i]);
  zb[i] = z;
}

// h_new = (1-z)*h + z*tanh(A3); emits f32 (output) + bf16 (GEMM operand)
__global__ void hnew_kernel(const float* __restrict__ A3, const float* __restrict__ zb,
                            const float* __restrict__ h, float* __restrict__ hout,
                            __bf16* __restrict__ hnbf, int n)
{
  int i = blockIdx.x * blockDim.x + threadIdx.x;
  if (i >= n) return;
  float ht = tanhf(A3[i]);
  float z  = zb[i];
  float hn = (1.0f - z) * h[i] + z * ht;
  hout[i] = hn;
  hnbf[i] = f2bf(hn);
}

// gate[b] = sigmoid(h_new[b] . Wg + bg)
__global__ __launch_bounds__(64)
void gate_kernel(const float* __restrict__ hn, const float* __restrict__ Wg,
                 const float* __restrict__ bg, float* __restrict__ gate)
{
  __shared__ float red[64];
  int t = threadIdx.x, b = blockIdx.x;
  float acc = 0.0f;
  for (int j = t; j < H_; j += 64) acc += hn[(size_t)b * H_ + j] * Wg[j];
  red[t] = acc;
  __syncthreads();
  for (int off = 32; off > 0; off >>= 1) {
    if (t < off) red[t] += red[t + off];
    __syncthreads();
  }
  if (t == 0) gate[b] = sigm(red[0] + bg[0]);
}

// Attention write per (b,m): second pass over memory (L2-resident)
__global__ __launch_bounds__(128)
void write_mem_kernel(const float* __restrict__ mem, const float* __restrict__ kq,
                      const float* __restrict__ vv, const float* __restrict__ rw,
                      const float* __restrict__ gate, float* __restrict__ out)
{
  __shared__ float tile[S_ * TSTR];
  __shared__ float red[S_];
  __shared__ float ks[D_];
  __shared__ float vs[D_];
  const int t = threadIdx.x;
  const int b = blockIdx.y;
  const int m = blockIdx.x;
  const float* mp = mem + (((size_t)b * M_ + m) * S_) * D_;
  if (t < D_) { ks[t] = kq[(size_t)b * D_ + t]; vs[t] = vv[(size_t)b * D_ + t]; }
  __syncthreads();
  const float* rowp = mp + (size_t)t * D_;
  float* trow = tile + t * TSTR;
  float acc = 0.0f;
  #pragma unroll
  for (int d = 0; d < D_; d += 4) {
    float4 v = *(const float4*)(rowp + d);
    trow[d] = v.x; trow[d+1] = v.y; trow[d+2] = v.z; trow[d+3] = v.w;
    acc += v.x*ks[d] + v.y*ks[d+1] + v.z*ks[d+2] + v.w*ks[d+3];
  }
  float sco = acc * 0.125f;
  red[t] = sco;
  __syncthreads();
  for (int off = 64; off > 0; off >>= 1) {
    if (t < off) red[t] = fmaxf(red[t], red[t + off]);
    __syncthreads();
  }
  float mx = red[0];
  __syncthreads();
  float p = __expf(sco - mx);
  red[t] = p;
  __syncthreads();
  for (int off = 64; off > 0; off >>= 1) {
    if (t < off) red[t] += red[t + off];
    __syncthreads();
  }
  float wgt = (rw[(size_t)b * M_ + m] * gate[b]) * p / red[0];
  float* op = out + (((size_t)b * M_ + m) * S_ + t) * (size_t)D_;
  #pragma unroll
  for (int d = 0; d < D_; d += 4) {
    float4 rr;
    rr.x = trow[d]   + wgt * vs[d];
    rr.y = trow[d+1] + wgt * vs[d+1];
    rr.z = trow[d+2] + wgt * vs[d+2];
    rr.w = trow[d+3] + wgt * vs[d+3];
    *(float4*)(op + d) = rr;
  }
}

// ---------------------------------------------------------------------------
extern "C" void kernel_launch(void* const* d_in, const int* in_sizes, int n_in,
                              void* d_out, int out_size, void* d_ws, size_t ws_size,
                              hipStream_t stream)
{
  const float* x    = (const float*)d_in[0];
  const float* h    = (const float*)d_in[1];
  const float* mem  = (const float*)d_in[2];
  const float* W_xr = (const float*)d_in[3];
  const float* W_hr = (const float*)d_in[4];
  const float* b_r  = (const float*)d_in[5];
  const float* W_xz = (const float*)d_in[6];
  const float* W_hz = (const float*)d_in[7];
  const float* b_z  = (const float*)d_in[8];
  const float* W_xh = (const float*)d_in[9];
  const float* W_hh = (const float*)d_in[10];
  const float* b_h  = (const float*)d_in[11];
  const float* Wr1  = (const float*)d_in[12];
  const float* br1  = (const float*)d_in[13];
  const float* Wr2  = (const float*)d_in[14];
  const float* br2  = (const float*)d_in[15];
  const float* Wq   = (const float*)d_in[16];
  const float* bq   = (const float*)d_in[17];
  const float* Wk   = (const float*)d_in[18];
  const float* bk   = (const float*)d_in[19];
  const float* Wv   = (const float*)d_in[20];
  const float* bv   = (const float*)d_in[21];
  const float* Wg   = (const float*)d_in[22];
  const float* bg   = (const float*)d_in[23];

  float* hout = (float*)d_out;                 // h_new: (B,H)
  float* mout = hout + (size_t)B_ * H_;        // memory_new: (B,M,S,D)

  // ---- workspace carve-up: f32 region then bf16 region (16B aligned) ----
  float* ws = (float*)d_ws;
  size_t o = 0;
  float* A1  = ws + o; o += (size_t)B_ * H_;
  float* A2  = ws + o; o += (size_t)B_ * H_;
  float* A3  = ws + o; o += (size_t)B_ * H_;
  float* ZB  = ws + o; o += (size_t)B_ * H_;
  float* L1  = ws + o; o += (size_t)B_ * RH_;
  float* Qb  = ws + o; o += (size_t)B_ * D_;
  float* CTX = ws + o; o += (size_t)B_ * D_;
  float* RW  = ws + o; o += (size_t)B_ * M_;
  float* KB  = ws + o; o += (size_t)B_ * D_;
  float* VB  = ws + o; o += (size_t)B_ * D_;
  float* GB  = ws + o; o += (size_t)B_;
  o = (o + 7) & ~(size_t)7;
  __bf16* bws = (__bf16*)(ws + o);
  size_t bo = 0;
  __bf16* xbf   = bws + bo; bo += (size_t)B_ * IN_;
  __bf16* hbf   = bws + bo; bo += (size_t)B_ * H_;
  __bf16* ctxbf = bws + bo; bo += (size_t)B_ * D_;
  __bf16* rhbf  = bws + bo; bo += (size_t)B_ * H_;
  __bf16* hnbf  = bws + bo; bo += (size_t)B_ * H_;
  __bf16* WtXR  = bws + bo; bo += (size_t)H_ * (IN_ + D_);   // Wt[N=H][K=576]
  __bf16* WtXZ  = bws + bo; bo += (size_t)H_ * (IN_ + D_);
  __bf16* WtXH  = bws + bo; bo += (size_t)H_ * (IN_ + D_);
  __bf16* WtHR  = bws + bo; bo += (size_t)H_ * H_;
  __bf16* WtHZ  = bws + bo; bo += (size_t)H_ * H_;
  __bf16* WtHH  = bws + bo; bo += (size_t)H_ * H_;
  __bf16* WtR1  = bws + bo; bo += (size_t)RH_ * (IN_ + H_);  // Wt[32][1536]
  __bf16* WtQ   = bws + bo; bo += (size_t)D_ * (IN_ + H_);   // Wt[64][1536]
  __bf16* WtK   = bws + bo; bo += (size_t)D_ * H_;
  __bf16* WtV   = bws + bo; bo += (size_t)D_ * H_;

  auto cvt = [&](const float* in, __bf16* out, int n) {
    cvt_bf16_kernel<<<(n / 4 + 255) / 256, 256, 0, stream>>>(in, out, n);
  };
  auto cvtT = [&](const float* W, __bf16* Wt, int K, int N) {
    cvtT_bf16_kernel<<<dim3(N / 32, K / 32), dim3(32, 8), 0, stream>>>(W, Wt, K, N);
  };
  // big GEMMs: N = H, wave tile 32x64, block covers 32x256
  auto gemmBig = [&](const __bf16* A, int lda, const __bf16* Bt, int ldk, int koff,
                     float* C, const float* bias, int K, bool acc) {
    dim3 g(H_ / 256, B_ / 32);
    if (acc) gemm_bf16_wmma<2, 4, true ><<<g, 128, 0, stream>>>(A, lda, Bt, ldk, koff, C, H_, bias, K);
    else     gemm_bf16_wmma<2, 4, false><<<g, 128, 0, stream>>>(A, lda, Bt, ldk, koff, C, H_, bias, K);
  };
  // small GEMMs: N in {32, 64}, one 16x16 wave tile per block
  auto gemmSmall = [&](const __bf16* A, int lda, const __bf16* Bt, int ldk, int koff,
                       float* C, int N, const float* bias, int K, bool acc) {
    dim3 g(N / 16, B_ / 16);
    if (acc) gemm_bf16_wmma<1, 1, true ><<<g, 32, 0, stream>>>(A, lda, Bt, ldk, koff, C, N, bias, K);
    else     gemm_bf16_wmma<1, 1, false><<<g, 32, 0, stream>>>(A, lda, Bt, ldk, koff, C, N, bias, K);
  };

  // ---- one-time bf16 conversion / weight transposition ----
  cvt(x, xbf, B_ * IN_);
  cvt(h, hbf, B_ * H_);
  cvtT(W_xr, WtXR, IN_ + D_, H_);
  cvtT(W_xz, WtXZ, IN_ + D_, H_);
  cvtT(W_xh, WtXH, IN_ + D_, H_);
  cvtT(W_hr, WtHR, H_, H_);
  cvtT(W_hz, WtHZ, H_, H_);
  cvtT(W_hh, WtHH, H_, H_);
  cvtT(Wr1,  WtR1, IN_ + H_, RH_);
  cvtT(Wq,   WtQ,  IN_ + H_, D_);
  cvtT(Wk,   WtK,  H_, D_);
  cvtT(Wv,   WtV,  H_, D_);
  zero_kernel<<<(B_ * D_ + 255) / 256, 256, 0, stream>>>(CTX, B_ * D_);

  // ---- ctx-independent GEMMs (xa=[x,ctx] and rin=[x,h] splits via koff) ----
  gemmBig(xbf, IN_, WtXR, IN_ + D_, 0, A1, b_r, IN_, false);
  gemmBig(hbf, H_,  WtHR, H_,       0, A1, nullptr, H_, true);
  gemmBig(xbf, IN_, WtXZ, IN_ + D_, 0, A2, b_z, IN_, false);
  gemmBig(hbf, H_,  WtHZ, H_,       0, A2, nullptr, H_, true);
  gemmBig(xbf, IN_, WtXH, IN_ + D_, 0, A3, b_h, IN_, false);
  gemmSmall(xbf, IN_, WtR1, IN_ + H_, 0,   L1, RH_, br1, IN_, false);
  gemmSmall(hbf, H_,  WtR1, IN_ + H_, IN_, L1, RH_, nullptr, H_, true);
  gemmSmall(xbf, IN_, WtQ,  IN_ + H_, 0,   Qb, D_, bq, IN_, false);
  gemmSmall(hbf, H_,  WtQ,  IN_ + H_, IN_, Qb, D_, nullptr, H_, true);

  // ---- router softmax, attention read -> ctx (bf16 copy for GEMM) ----
  router_kernel<<<(B_ + 127) / 128, 128, 0, stream>>>(L1, Wr2, br2, RW);
  read_ctx_kernel<<<dim3(M_, B_), 128, 0, stream>>>(mem, Qb, RW, CTX);
  cvt(CTX, ctxbf, B_ * D_);

  // ---- ctx contributions (K = D = 64, koff = IN) ----
  gemmBig(ctxbf, D_, WtXR, IN_ + D_, IN_, A1, nullptr, D_, true);
  gemmBig(ctxbf, D_, WtXZ, IN_ + D_, IN_, A2, nullptr, D_, true);
  gemmBig(ctxbf, D_, WtXH, IN_ + D_, IN_, A3, nullptr, D_, true);

  // ---- GRU ----
  gru_gates_kernel<<<(B_ * H_ + 255) / 256, 256, 0, stream>>>(A1, A2, h, rhbf, ZB, B_ * H_);
  gemmBig(rhbf, H_, WtHH, H_, 0, A3, nullptr, H_, true);
  hnew_kernel<<<(B_ * H_ + 255) / 256, 256, 0, stream>>>(A3, ZB, h, hout, hnbf, B_ * H_);

  // ---- k / v / gate ----
  gemmSmall(hnbf, H_, WtK, H_, 0, KB, D_, bk, H_, false);
  gemmSmall(hnbf, H_, WtV, H_, 0, VB, D_, bv, H_, false);
  gate_kernel<<<B_, 64, 0, stream>>>(hout, Wg, bg, GB);

  // ---- attention write -> memory_new ----
  write_mem_kernel<<<dim3(M_, B_), 128, 0, stream>>>(mem, KB, VB, RW, GB, mout);
}
